// down_block_2516850835581
// MI455X (gfx1250) — matrix operand order repaired and checked by reading, not compiled, not run on hardware
//
#include <hip/hip_runtime.h>

// ---------------------------------------------------------------------------
// Problem constants (fixed by the reference)
// ---------------------------------------------------------------------------
constexpr int N_IN   = 163842;
constexpr int N_OUT  = 40962;
constexpr int IN_CH  = 128;
constexpr int OUT_CH = 256;
constexpr float BN_EPS    = 1e-5f;
constexpr float NEG_SLOPE = 0.2f;

// ---------------------------------------------------------------------------
// WMMA types (gfx1250, wave32)
// ---------------------------------------------------------------------------
typedef __attribute__((ext_vector_type(16))) __bf16 v16bf;
typedef __attribute__((ext_vector_type(8)))  float  v8f;

struct alignas(16) U4 { unsigned int w[4]; };      // 128-bit chunk
union ABFrag { v16bf v; U4 q[2]; };                // 8 VGPRs per lane

// ---------------------------------------------------------------------------
// CDNA5 async global->LDS copies (ASYNCcnt-tracked, bypass VGPRs)
// ---------------------------------------------------------------------------
__device__ __forceinline__ void async_b128(unsigned lds_off, const void* gaddr) {
    asm volatile("global_load_async_to_lds_b128 %0, %1, off"
                 :: "v"(lds_off), "v"(gaddr) : "memory");
}
__device__ __forceinline__ void async_b128_x2(unsigned lds_off, const void* gaddr) {
    // IOFFSET applies to BOTH the LDS and global address (ISA 08_async_tensor §4.4)
    asm volatile("global_load_async_to_lds_b128 %0, %1, off\n\t"
                 "global_load_async_to_lds_b128 %0, %1, off offset:16"
                 :: "v"(lds_off), "v"(gaddr) : "memory");
}
__device__ __forceinline__ void wait_async_le3() {
    asm volatile("s_wait_asynccnt 0x3" ::: "memory");
}
__device__ __forceinline__ void wait_async_0() {
    asm volatile("s_wait_asynccnt 0x0" ::: "memory");
}

// ---------------------------------------------------------------------------
// Mean pooling: xp[m][c] = mean_j x[pidx[7m+j]][c], emitted as bf16
// ---------------------------------------------------------------------------
__global__ void pool_kernel(const float* __restrict__ x,
                            const int* __restrict__ pidx,
                            __bf16* __restrict__ xp, int M) {
    int i = blockIdx.x * blockDim.x + threadIdx.x;
    if (i >= M * IN_CH) return;
    int m = i >> 7;          // /128
    int c = i & (IN_CH - 1);
    float s = 0.f;
#pragma unroll
    for (int j = 0; j < 7; ++j) {
        int idx = pidx[7 * m + j];
        s += x[(size_t)idx * IN_CH + c];
    }
    xp[i] = (__bf16)(s * (1.0f / 7.0f));
}

// ---------------------------------------------------------------------------
// Weight transpose + f32->bf16: wT[n][k] = W[k][n]   (W is [K][256] row-major)
// ---------------------------------------------------------------------------
__global__ void wtrans_kernel(const float* __restrict__ W,
                              __bf16* __restrict__ wT, int K) {
    int i = blockIdx.x * blockDim.x + threadIdx.x;
    if (i >= K * OUT_CH) return;
    int k = i >> 8;           // /256
    int n = i & (OUT_CH - 1);
    wT[(size_t)n * K + k] = (__bf16)W[i];   // W[k*256+n]
}

// ---------------------------------------------------------------------------
// Gather-GEMM with WMMA + async double-buffered LDS staging.
//   out[M,256] = concat7(Asrc[nbr])[M, 7*CH] @ W + bias
// Block: 512 threads = 16 waves. 128 rows x 256 cols per block.
//   wave -> (wm = wave>>1 in 0..7 : 16-row strip, wn = wave&1 : 128-col strip)
// Per K32 step each thread issues 3 async b128 copies (1 A + 2 B).
// ---------------------------------------------------------------------------
template <int CH>
__global__ void __launch_bounds__(512)
gemm_gather_wmma(const __bf16* __restrict__ Asrc,
                 const int* __restrict__ nbr,
                 const __bf16* __restrict__ wT,
                 const float* __restrict__ bias,
                 float* __restrict__ out, int M) {
    constexpr int K     = 7 * CH;
    constexpr int NK    = K / 32;
    constexpr int JSTEP = CH / 32;           // K32 tiles per neighbor slot

    __shared__ alignas(16) __bf16 ldsA[2][128 * 32];   // 2 x 8 KB gathered A
    __shared__ alignas(16) __bf16 ldsB[2][256 * 32];   // 2 x 16 KB  W^T tile

    const int tid  = threadIdx.x;
    const int lane = tid & 31;
    const int wave = tid >> 5;
    const int wm   = wave >> 1;       // 0..7 -> 16-row strip
    const int wn   = wave & 1;        // 0..1 -> 128-col strip
    const int lh   = lane >> 4;       // lane half
    const int lm   = lane & 15;

    const int row0 = blockIdx.x * 128;

    // ---- staging assignments ------------------------------------------------
    // A: thread -> (row ar = tid>>2 in 0..127, 16B segment aseg = tid&3)
    const int ar   = tid >> 2;
    const int aseg = tid & 3;
    int am = row0 + ar; if (am >= M) am = M - 1;    // clamp; stores masked later
    // B: thread -> (N-row bn = tid>>1 in 0..255, 32B half bh = tid&1)
    const int bnr = tid >> 1;
    const int bhf = tid & 1;
    const __bf16* bsrc0 = wT + (size_t)bnr * K + bhf * 16;

    const unsigned aoff0 = (unsigned)(size_t)&ldsA[0][ar * 32 + aseg * 8];
    const unsigned boff0 = (unsigned)(size_t)&ldsB[0][bnr * 32 + bhf * 16];
    constexpr unsigned ABUF = sizeof(__bf16) * 128 * 32;   // 8192
    constexpr unsigned BBUF = sizeof(__bf16) * 256 * 32;   // 16384

    auto issue = [&](int ks, int buf) {
        const int k0 = ks * 32;
        const int j  = ks / JSTEP;
        const int c0 = k0 - j * CH;
        const int idx = nbr[7 * am + j];                       // cached gather idx
        async_b128(aoff0 + (unsigned)buf * ABUF,
                   Asrc + (size_t)idx * CH + c0 + aseg * 8);
        async_b128_x2(boff0 + (unsigned)buf * BBUF, bsrc0 + k0);
    };

    v8f acc[8] = {};                  // 8 tiles of 16x16 f32 per wave

    issue(0, 0);
    for (int ks = 0; ks < NK; ++ks) {
        const int buf = ks & 1;
        if (ks + 1 < NK) {            // prefetch next tile into other buffer
            issue(ks + 1, buf ^ 1);
            wait_async_le3();         // in-order: current tile's 3 copies done
        } else {
            wait_async_0();
        }
        __syncthreads();

        const __bf16* At = &ldsA[0][0] + buf * (128 * 32);
        const __bf16* Bt = &ldsB[0][0] + buf * (256 * 32);

        // A fragment (16x32 bf16):
        //   lane half 0: elems 0..7 -> K 0..7,  elems 8..15 -> K 16..23
        //   lane half 1: elems 0..7 -> K 8..15, elems 8..15 -> K 24..31
        ABFrag a;
        {
            const __bf16* ap = At + (wm * 16 + lm) * 32 + lh * 8;
            a.q[0] = *(const U4*)(ap);
            a.q[1] = *(const U4*)(ap + 16);
        }
#pragma unroll
        for (int nt = 0; nt < 8; ++nt) {
            ABFrag b;   // B (32x16): N = lm, K same as A fragment (8*lh + e, 16 + 8*lh + e-8)
            const __bf16* bp = Bt + (wn * 128 + nt * 16 + lm) * 32 + lh * 8;
            b.q[0] = *(const U4*)(bp);
            b.q[1] = *(const U4*)(bp + 16);
            acc[nt] = __builtin_amdgcn_wmma_f32_16x16x32_bf16(
                false, a.v, false, b.v, (short)0, acc[nt], false, false);
        }
        __syncthreads();              // protects buf from tile ks+2 prefetch overwrite
    }

    // ---- store D + bias. C/D layout: VGPR r -> M = r + 8*lh, N = lm
    const int colb = wn * 128;
#pragma unroll
    for (int nt = 0; nt < 8; ++nt) {
        const int col = colb + nt * 16 + lm;
        const float bv = bias[col];
#pragma unroll
        for (int r = 0; r < 8; ++r) {
            const int row = row0 + wm * 16 + lh * 8 + r;
            if (row < M)
                out[(size_t)row * OUT_CH + col] = acc[nt][r] + bv;
        }
    }
}

// ---------------------------------------------------------------------------
// BatchNorm column statistics: partial sums over a 256-row chunk per block,
// thread t owns channel t (fully coalesced), atomically merged.
// ---------------------------------------------------------------------------
__global__ void bn_stats_kernel(const float* __restrict__ h, int M,
                                float* __restrict__ sum,
                                float* __restrict__ ssum) {
    const int c  = threadIdx.x;            // 0..255 channel
    const int r0 = blockIdx.x * 256;
    int rend = r0 + 256; if (rend > M) rend = M;
    float s = 0.f, q = 0.f;
    for (int r = r0; r < rend; ++r) {
        float v = h[(size_t)r * OUT_CH + c];
        s += v; q += v * v;
    }
    atomicAdd(&sum[c], s);
    atomicAdd(&ssum[c], q);
}

__global__ void zero_kernel(float* p, int n) {
    int i = blockIdx.x * blockDim.x + threadIdx.x;
    if (i < n) p[i] = 0.f;
}

// ---------------------------------------------------------------------------
// Fused BN(training stats) + LeakyReLU; OutT selects bf16 (feeds next gather)
// or f32 (final output).
// ---------------------------------------------------------------------------
template <typename OutT>
__global__ void bn_apply_kernel(const float* __restrict__ h,
                                const float* __restrict__ sum,
                                const float* __restrict__ ssum,
                                const float* __restrict__ gamma,
                                const float* __restrict__ beta,
                                OutT* __restrict__ out, int M) {
    int i = blockIdx.x * blockDim.x + threadIdx.x;
    if (i >= M * OUT_CH) return;
    int c = i & (OUT_CH - 1);
    float inv  = 1.0f / (float)M;
    float mean = sum[c] * inv;
    float var  = ssum[c] * inv - mean * mean;
    float sc   = gamma[c] * rsqrtf(var + BN_EPS);
    float sh   = beta[c] - mean * sc;
    float y = h[i] * sc + sh;
    y = (y < 0.f) ? NEG_SLOPE * y : y;
    out[i] = (OutT)y;
}

// ---------------------------------------------------------------------------
// Launcher
// ---------------------------------------------------------------------------
extern "C" void kernel_launch(void* const* d_in, const int* in_sizes, int n_in,
                              void* d_out, int out_size, void* d_ws, size_t ws_size,
                              hipStream_t stream) {
    const float* x     = (const float*)d_in[0];
    const int*   pidx  = (const int*)  d_in[1];
    const int*   nidx  = (const int*)  d_in[2];
    const float* W1    = (const float*)d_in[3];
    const float* b1    = (const float*)d_in[4];
    const float* g1    = (const float*)d_in[5];
    const float* be1   = (const float*)d_in[6];
    const float* W2    = (const float*)d_in[7];
    const float* b2    = (const float*)d_in[8];
    const float* g2    = (const float*)d_in[9];
    const float* be2   = (const float*)d_in[10];
    float* out = (float*)d_out;

    constexpr int K1 = 7 * IN_CH;    // 896
    constexpr int K2 = 7 * OUT_CH;   // 1792
    const int M = N_OUT;

    // ---- workspace carve-up (256B aligned slices)
    char* ws = (char*)d_ws;
    size_t off = 0;
    auto take = [&](size_t bytes) -> char* {
        char* p = ws + off;
        off = (off + bytes + 255) & ~(size_t)255;
        return p;
    };
    __bf16* xp    = (__bf16*)take((size_t)M * IN_CH * 2);        // 10.5 MB
    __bf16* w1T   = (__bf16*)take((size_t)K1 * OUT_CH * 2);      // 0.46 MB
    __bf16* w2T   = (__bf16*)take((size_t)K2 * OUT_CH * 2);      // 0.92 MB
    float*  hraw  = (float*) take((size_t)M * OUT_CH * 4);       // 42 MB (reused for h2)
    __bf16* hbf   = (__bf16*)take((size_t)M * OUT_CH * 2);       // 21 MB
    float*  stats = (float*) take(4 * OUT_CH * 4);               // sum1,ss1,sum2,ss2
    float* sum1 = stats;
    float* ss1  = stats + OUT_CH;
    float* sum2 = stats + 2 * OUT_CH;
    float* ss2  = stats + 3 * OUT_CH;
    (void)ws_size; (void)n_in; (void)in_sizes; (void)out_size;

    // ---- 0. zero BN accumulators (graph-replay safe: done every call)
    zero_kernel<<<4, 256, 0, stream>>>(stats, 4 * OUT_CH);

    // ---- 1. mean pooling -> xp (bf16)
    {
        int total = M * IN_CH;
        pool_kernel<<<(total + 255) / 256, 256, 0, stream>>>(x, pidx, xp, M);
    }

    // ---- 2. weight transpose+convert
    wtrans_kernel<<<(K1 * OUT_CH + 255) / 256, 256, 0, stream>>>(W1, w1T, K1);
    wtrans_kernel<<<(K2 * OUT_CH + 255) / 256, 256, 0, stream>>>(W2, w2T, K2);

    // ---- 3. GEMM1 (gather 7x128 -> 896) via WMMA bf16, async double-buffer
    {
        dim3 grid((M + 127) / 128);
        gemm_gather_wmma<IN_CH><<<grid, 512, 0, stream>>>(xp, nidx, w1T, b1, hraw, M);
    }

    // ---- 4. BN1 stats + apply -> h (bf16)
    bn_stats_kernel<<<(M + 255) / 256, 256, 0, stream>>>(hraw, M, sum1, ss1);
    bn_apply_kernel<__bf16><<<(M * OUT_CH + 255) / 256, 256, 0, stream>>>(
        hraw, sum1, ss1, g1, be1, hbf, M);

    // ---- 5. GEMM2 (gather 7x256 -> 1792) via WMMA bf16, reuse hraw for h2
    {
        dim3 grid((M + 127) / 128);
        gemm_gather_wmma<OUT_CH><<<grid, 512, 0, stream>>>(hbf, nidx, w2T, b2, hraw, M);
    }

    // ---- 6. BN2 stats + apply -> out (f32)
    bn_stats_kernel<<<(M + 255) / 256, 256, 0, stream>>>(hraw, M, sum2, ss2);
    bn_apply_kernel<float><<<(M * OUT_CH + 255) / 256, 256, 0, stream>>>(
        hraw, sum2, ss2, g2, be2, out, M);
}
